// _Feature_layer_48507360641428
// MI455X (gfx1250) — matrix operand order repaired and checked by reading, not compiled, not run on hardware
//
#include <hip/hip_runtime.h>
#include <hip/hip_bf16.h>

typedef unsigned short ushort_t;
typedef __attribute__((ext_vector_type(16))) __bf16 v16bf;
typedef __attribute__((ext_vector_type(8)))  __bf16 v8bf;
typedef __attribute__((ext_vector_type(8)))  float  v8f;
typedef __attribute__((ext_vector_type(4)))  unsigned int v4u;
typedef __attribute__((ext_vector_type(8)))  int v8i;
typedef __attribute__((ext_vector_type(4)))  int v4i;

#if defined(__gfx1250__) && __has_builtin(__builtin_amdgcn_tensor_load_to_lds)
#define HAVE_TDM 1
#else
#define HAVE_TDM 0
#endif

__device__ __forceinline__ float bf2f(ushort_t u) {
    unsigned int x = ((unsigned int)u) << 16;
    float f;
    __builtin_memcpy(&f, &x, 4);
    return f;
}
__device__ __forceinline__ ushort_t f2bf(float f) {
    unsigned int x;
    __builtin_memcpy(&x, &f, 4);
    unsigned int r = x + 0x7FFFu + ((x >> 16) & 1u);   // round-to-nearest-even
    return (ushort_t)(r >> 16);
}

// ---------------------------------------------------------------------------
// Weight repack: OIHW f32 -> [tap][cout][cin] bf16  (A-matrix friendly layout)
// ---------------------------------------------------------------------------
__global__ __launch_bounds__(256)
void wprep(const float* __restrict__ w, ushort_t* __restrict__ dst,
           int Cin, int Cout, int total) {
    int idx = blockIdx.x * 256 + threadIdx.x;
    if (idx >= total) return;
    int tap  = idx % 9;
    int t    = idx / 9;
    int cin  = t % Cin;
    int cout = t / Cin;
    dst[((size_t)tap * Cout + cout) * Cin + cin] = f2bf(w[idx]);
}

// ---------------------------------------------------------------------------
// BN fold: scale = gamma*rsqrt(var+eps), bias = (b-mean)*scale + beta
// ---------------------------------------------------------------------------
__global__ __launch_bounds__(256)
void sbprep(const float* __restrict__ b, const float* __restrict__ gamma,
            const float* __restrict__ beta, const float* __restrict__ mean,
            const float* __restrict__ var, float* __restrict__ scale,
            float* __restrict__ bias, int C) {
    int c = blockIdx.x * 256 + threadIdx.x;
    if (c >= C) return;
    float inv = gamma[c] * rsqrtf(var[c] + 1e-5f);
    scale[c] = inv;
    bias[c]  = (b[c] - mean[c]) * inv + beta[c];
}

// ---------------------------------------------------------------------------
// Zero the 1-pixel halo border of a padded (H+2)x(W+2)xC bf16 buffer.
// ---------------------------------------------------------------------------
__global__ __launch_bounds__(256)
void zero_halo(ushort_t* __restrict__ buf, int H, int W, int C) {
    int idx = blockIdx.x * 256 + threadIdx.x;
    int cg  = C >> 3;                       // 8 bf16 per thread
    int Wp  = W + 2;
    int total = (2 * Wp + 2 * H) * cg;
    if (idx >= total) return;
    int pix = idx / cg;
    int c8  = (idx % cg) * 8;
    int row, col;
    if (pix < Wp)            { row = 0;     col = pix; }
    else if (pix < 2 * Wp)   { row = H + 1; col = pix - Wp; }
    else {
        int q = pix - 2 * Wp;
        row = 1 + (q >> 1);
        col = (q & 1) ? (W + 1) : 0;
    }
    uint4 z = make_uint4(0u, 0u, 0u, 0u);
    *(uint4*)(buf + ((size_t)row * Wp + col) * C + c8) = z;
}

// ---------------------------------------------------------------------------
// Layer 0: Cin=3 direct conv (K=27, not WMMA-tileable; ~0.6% of total FLOPs).
// Input NCHW f32, output halo-padded HWC bf16, fused BN+ReLU.
// ---------------------------------------------------------------------------
__global__ __launch_bounds__(256)
void conv0_direct(const float* __restrict__ img, ushort_t* __restrict__ out,
                  const float* __restrict__ w, const float* __restrict__ scale,
                  const float* __restrict__ bias) {
    int idx  = blockIdx.x * 256 + threadIdx.x;     // 256*256*64 threads
    int cout = idx & 63;
    int s    = idx >> 6;
    int y    = s >> 8;
    int x    = s & 255;
    float acc = 0.f;
#pragma unroll
    for (int ch = 0; ch < 3; ++ch) {
#pragma unroll
        for (int ky = 0; ky < 3; ++ky) {
            int yy = y + ky - 1;
            if (yy < 0 || yy > 255) continue;
#pragma unroll
            for (int kx = 0; kx < 3; ++kx) {
                int xx = x + kx - 1;
                if (xx < 0 || xx > 255) continue;
                acc += img[ch * 65536 + yy * 256 + xx] *
                       w[((cout * 3 + ch) * 3 + ky) * 3 + kx];
            }
        }
    }
    float v = acc * scale[cout] + bias[cout];
    v = v > 0.f ? v : 0.f;
    out[((size_t)(y + 1) * 258 + (x + 1)) * 64 + cout] = f2bf(v);
}

// ---------------------------------------------------------------------------
// TDM descriptor issue: 3-D tile (32 cin x 16 px x 4 rows) global -> LDS.
// D# packing per CDNA5 ISA ch.8 (group0..3). Tensor dims set huge: the
// halo-padded layout guarantees every tap read is in-bounds.
// LDS allocation base assumed 0 for the single __shared__ array.
// ---------------------------------------------------------------------------
#if HAVE_TDM
__device__ __forceinline__ void tdm_load_tile(const ushort_t* gsrc,
                                              unsigned ldsByte, int Cin,
                                              int rowStrideElems) {
    unsigned long long ga = (unsigned long long)gsrc;
    v4u g0;
    g0.x = 1u;                                        // count=1, user mode
    g0.y = ldsByte;                                   // lds_addr [63:32]
    g0.z = (unsigned)(ga & 0xffffffffu);              // global_addr [95:64]
    g0.w = (unsigned)((ga >> 32) & 0x01ffffffu)       // global_addr [120:96]
         | (2u << 30);                                // type=2 ("image")
    unsigned td0 = 0x7fffffffu, td1 = 0x7fffffffu;    // no OOB clipping needed
    unsigned s0  = (unsigned)Cin;                     // dim0 stride (elems)
    unsigned s1  = (unsigned)rowStrideElems;          // dim1 stride (elems)
    v8i g1;
    g1[0] = (int)(1u << 16);                          // data_size=1 (2 bytes)
    g1[1] = (int)((td0 & 0xffffu) << 16);             // tensor_dim0 lo16
    g1[2] = (int)(((td0 >> 16) & 0xffffu) | ((td1 & 0xffffu) << 16));
    g1[3] = (int)(((td1 >> 16) & 0xffffu) | (32u << 16)); // tile_dim0=32
    g1[4] = (int)(16u | (4u << 16));                  // tile_dim1=16, tile_dim2=4
    g1[5] = (int)s0;                                  // dim0 stride lo32
    g1[6] = (int)((s1 & 0xffffu) << 16);              // dim0 hi16=0 | dim1 lo16
    g1[7] = (int)(s1 >> 16);                          // dim1 stride hi
    v4i g2;
    g2[0] = 0x7fffffff;                               // tensor_dim2 (huge)
    g2[1] = 0;                                        // tensor_dim3
    g2[2] = 0;                                        // dim2 stride (unused)
    g2[3] = 0;                                        // tile_dim3=0
    v4i g3 = {0, 0, 0, 0};
#if __clang_major__ >= 23
    // amdgpu-toolchain / therock-10.0: 6-arg form (extra descriptor group)
    v8i g4 = {0, 0, 0, 0, 0, 0, 0, 0};
    __builtin_amdgcn_tensor_load_to_lds(g0, g1, g2, g3, g4, 0);
#else
    // ROCm 7.2 / clang-22: 5-arg form
    __builtin_amdgcn_tensor_load_to_lds(g0, g1, g2, g3, 0);
#endif
}
#endif

// ---------------------------------------------------------------------------
// Implicit-GEMM 3x3 SAME conv via v_wmma_f32_16x16x32_bf16.
//   A = weight tile 16x32 (rows=cout, contiguous cin in [tap][cout][cin])
//   B = activation tile 32xN from LDS (staged by TDM, double buffered)
//   Block: 8 waves = 4 cout-tiles x 2 spatial-halves, 2 WMMAs/wave/step
//   Block output tile: 64 cout x 64 spatial (16-wide x 4-tall patch).
// Activations are halo-padded HWC bf16: no boundary branches anywhere.
//
// TDM writes LDS through a descriptor-encoded integer address the compiler
// cannot alias to `smem`. Two things are required so clang keeps the LDS
// fragment loads: (1) ESCAPE the smem pointer into an asm operand once, and
// (2) a memory-clobber asm each iteration. Without the escape, smem is a
// non-captured allocation with no visible stores, so loads fold to undef.
// ---------------------------------------------------------------------------
__global__ __launch_bounds__(256)
void conv3x3_wmma_bf16(const ushort_t* __restrict__ actIn,
                       ushort_t* __restrict__ actOut,
                       const ushort_t* __restrict__ wt,   // [9][Cout][Cin] bf16
                       const float* __restrict__ scale,
                       const float* __restrict__ bias,
                       int H, int W, int Cin, int Cout) {
    __shared__ __align__(128) ushort_t smem[2][64 * 32];  // 2 x 4KB tiles

    const int lane  = threadIdx.x & 31;
    const int wave  = threadIdx.x >> 5;
    const int mtile = wave & 3;          // cout tile (0..3)
    const int ntile = wave >> 2;         // spatial half (0..1)
    const int lrow  = lane & 15;
    const int khalf = lane >> 4;

    const int pxw = W >> 4;
    const int px  = blockIdx.x % pxw;
    const int py  = blockIdx.x / pxw;
    const int x0  = px << 4;             // patch origin (16 wide, 4 tall)
    const int y0  = py << 2;
    const int coutBlk = blockIdx.y << 6;

    const int Wp        = W + 2;
    const int rowStride = Wp * Cin;
    const int kSteps    = Cin >> 5;
    const int S         = 9 * kSteps;

    v8f acc[2];
    acc[0] = (v8f){0.f, 0.f, 0.f, 0.f, 0.f, 0.f, 0.f, 0.f};
    acc[1] = (v8f){0.f, 0.f, 0.f, 0.f, 0.f, 0.f, 0.f, 0.f};

    // Escape the LDS pointer: after this, memory-clobber asms (and the TDM)
    // may alias smem, so its loads can no longer be folded away.
    asm volatile("" ::"v"(&smem[0][0]) : "memory");

#if HAVE_TDM
    auto issue = [&](int step, int bufsel) {
        int tap = step / kSteps;
        int k0  = (step - tap * kSteps) << 5;
        int dyt = tap / 3, dxt = tap % 3;
        const ushort_t* gsrc =
            actIn + ((size_t)(y0 + dyt) * Wp + (x0 + dxt)) * Cin + k0;
        tdm_load_tile(gsrc, (unsigned)(bufsel * 64 * 32 * 2), Cin, rowStride);
    };
    if (wave == 0) issue(0, 0);
#endif

    for (int step = 0; step < S; ++step) {
#if HAVE_TDM
        const int bufsel = step & 1;
        if (wave == 0) {
            if (step + 1 < S) {
                issue(step + 1, bufsel ^ 1);                 // prefetch next
                __builtin_amdgcn_s_wait_tensorcnt((short)1); // current done
            } else {
                __builtin_amdgcn_s_wait_tensorcnt((short)0);
            }
        }
        __syncthreads();
        // TDM wrote LDS behind the compiler's back: with smem escaped above,
        // this clobber forces the ds_loads below to be (re)issued.
        asm volatile("" ::: "memory");
#else
        const int bufsel = 0;
        {   // manual staging fallback: 256 threads x 8 bf16 = 4KB tile
            int tap = step / kSteps;
            int k0  = (step - tap * kSteps) << 5;
            int dyt = tap / 3, dxt = tap % 3;
            int sl  = threadIdx.x >> 2;            // 0..63 tile row
            int cg  = (threadIdx.x & 3) * 8;       // cin sub-offset
            int rr  = sl >> 4, pp = sl & 15;
            const ushort_t* src =
                actIn + ((size_t)(y0 + dyt + rr) * Wp + (x0 + dxt + pp)) * Cin +
                k0 + cg;
            __syncthreads();
            *(uint4*)&smem[0][sl * 32 + cg] = *(const uint4*)src;
            __syncthreads();
        }
#endif
        // A fragment (weights) from global (L2-resident):
        // lane holds row M=lrow, K = {khalf*16..+15}
        {
            int tap = step / kSteps;
            int k0  = (step - tap * kSteps) << 5;
            const ushort_t* wp =
                wt + (((size_t)tap * Cout + coutBlk + mtile * 16 + lrow) * Cin +
                      k0 + khalf * 16);
            __builtin_prefetch(wp + 32, 0, 1);
            v8bf alo = *(const v8bf*)(wp);
            v8bf ahi = *(const v8bf*)(wp + 8);
            v16bf a = __builtin_shufflevector(alo, ahi, 0, 1, 2, 3, 4, 5, 6, 7,
                                              8, 9, 10, 11, 12, 13, 14, 15);

            const ushort_t* sb = &smem[bufsel][0];
            const v16bf b0 =
                *(const v16bf*)(sb + ((ntile * 32 + lrow) * 32 + khalf * 16));
            const v16bf b1 = *(const v16bf*)(sb + ((ntile * 32 + 16 + lrow) * 32 +
                                                   khalf * 16));
            acc[0] = __builtin_amdgcn_wmma_f32_16x16x32_bf16(
                false, a, false, b0, (short)0, acc[0], false, false);
            acc[1] = __builtin_amdgcn_wmma_f32_16x16x32_bf16(
                false, a, false, b1, (short)0, acc[1], false, false);
        }
#if HAVE_TDM
        __syncthreads();   // all waves done with buf before TDM overwrites it
#endif
    }

    // Epilogue: fused BN + ReLU, bf16 convert, halo-padded HWC store.
    const int cbase = coutBlk + mtile * 16 + khalf * 8;
#pragma unroll
    for (int nsub = 0; nsub < 2; ++nsub) {
        const int n  = ntile * 32 + nsub * 16 + lrow;  // tile-local spatial
        const int yy = y0 + (n >> 4);
        const int xx = x0 + (n & 15);
        __align__(16) ushort_t outv[8];
#pragma unroll
        for (int r = 0; r < 8; ++r) {
            const int c = cbase + r;
            float v = acc[nsub][r] * scale[c] + bias[c];
            v = v > 0.f ? v : 0.f;
            outv[r] = f2bf(v);
        }
        *(uint4*)(actOut + ((size_t)(yy + 1) * Wp + (xx + 1)) * Cout + cbase) =
            *(const uint4*)outv;
    }
}

// ---------------------------------------------------------------------------
// 2x2 max pool, halo-padded HWC bf16 in and out (4 channels per thread).
// ---------------------------------------------------------------------------
__global__ __launch_bounds__(256)
void maxpool2x2(const ushort_t* __restrict__ in, ushort_t* __restrict__ out,
                int W2, int C) {
    int idx = blockIdx.x * 256 + threadIdx.x;
    int cg4 = C >> 2;
    int c4  = (idx % cg4) * 4;
    int s   = idx / cg4;
    int x   = s % W2;
    int y   = s / W2;
    const int WpIn = 2 * W2 + 2;
    const ushort_t* p00 =
        in + ((size_t)(2 * y + 1) * WpIn + (2 * x + 1)) * C + c4;
    const ushort_t* p01 = p00 + C;
    const ushort_t* p10 = p00 + (size_t)WpIn * C;
    const ushort_t* p11 = p10 + C;
    __align__(8) ushort_t o[4];
#pragma unroll
    for (int i = 0; i < 4; ++i) {
        float m = fmaxf(fmaxf(bf2f(p00[i]), bf2f(p01[i])),
                        fmaxf(bf2f(p10[i]), bf2f(p11[i])));
        o[i] = f2bf(m);
    }
    *(uint2*)(out + ((size_t)(y + 1) * (W2 + 2) + (x + 1)) * C + c4) =
        *(const uint2*)o;
}

// ---------------------------------------------------------------------------
// ROI adaptive max pool: feat 18x18x512 halo-padded HWC bf16
// -> out (128,512,7,7) f32. Mirrors the reference bin arithmetic.
// ---------------------------------------------------------------------------
__global__ __launch_bounds__(256)
void roi_pool_k(const ushort_t* __restrict__ feat,
                const float* __restrict__ rois,
                float* __restrict__ out) {
    const int n = blockIdx.x;
    const float* r = rois + n * 4;
    int xi = (int)ceilf(r[0] * 0.0625f);
    int yi = (int)ceilf(r[1] * 0.0625f);
    int wi = (int)ceilf(r[2] * 0.0625f);
    int hi = (int)ceilf(r[3] * 0.0625f);
    wi = max(wi, 7);
    hi = max(hi, 7);
    int x1   = min(xi, max(16 - wi, 0));
    int y1   = min(yi, max(16 - hi, 0));
    int weff = min(wi, 16);
    int heff = min(hi, 16);

    for (int idx = threadIdx.x; idx < 512 * 49; idx += 256) {
        int c  = idx / 49;
        int ij = idx % 49;
        int i  = ij / 7, j = ij % 7;
        int sc = (j * weff) / 7, ec = ((j + 1) * weff + 6) / 7;
        int sr = (i * heff) / 7, er = ((i + 1) * heff + 6) / 7;
        float m = -3.4e38f;
        for (int rr = sr; rr < er; ++rr) {
            int row = min(max(y1 + rr, 0), 15);
            for (int cc = sc; cc < ec; ++cc) {
                int col = min(max(x1 + cc, 0), 15);
                m = fmaxf(m, bf2f(feat[((size_t)(row + 1) * 18 + (col + 1)) * 512 + c]));
            }
        }
        out[(((size_t)n * 512 + c) * 7 + i) * 7 + j] = m;
    }
}

// ---------------------------------------------------------------------------
// Host orchestration
// ---------------------------------------------------------------------------
extern "C" void kernel_launch(void* const* d_in, const int* in_sizes, int n_in,
                              void* d_out, int out_size, void* d_ws,
                              size_t ws_size, hipStream_t stream) {
    (void)in_sizes; (void)n_in; (void)out_size; (void)ws_size;

    const float* img  = (const float*)d_in[0];
    const float* rois = (const float*)d_in[1];

    static const int CO[13] = {64, 64, 128, 128, 256, 256, 256,
                               512, 512, 512, 512, 512, 512};
    static const int CI[13] = {3, 64, 64, 128, 128, 256, 256,
                               256, 512, 512, 512, 512, 512};
    // pools occur after conv layers 1, 3, 6, 9 (0-based)

    char* ws = (char*)d_ws;
    size_t off = 0;
    auto carve = [&](size_t bytes) -> void* {
        void* p = ws + off;
        off = (off + bytes + 255) & ~(size_t)255;
        return p;
    };

    ushort_t* wt[13] = {nullptr};
    for (int l = 1; l < 13; ++l)
        wt[l] = (ushort_t*)carve((size_t)CO[l] * CI[l] * 9 * sizeof(ushort_t));
    float* scl[13];
    float* bia[13];
    for (int l = 0; l < 13; ++l) {
        scl[l] = (float*)carve((size_t)CO[l] * sizeof(float));
        bia[l] = (float*)carve((size_t)CO[l] * sizeof(float));
    }
    // halo-padded activation ping-pong: (256+2)^2 x 64 bf16 max
    ushort_t* actA = (ushort_t*)carve((size_t)258 * 258 * 64 * sizeof(ushort_t));
    ushort_t* actB = (ushort_t*)carve((size_t)258 * 258 * 64 * sizeof(ushort_t));

    // ---- parameter prep ----
    for (int l = 0; l < 13; ++l) {
        const float* w     = (const float*)d_in[2 + 6 * l + 0];
        const float* b     = (const float*)d_in[2 + 6 * l + 1];
        const float* gamma = (const float*)d_in[2 + 6 * l + 2];
        const float* beta  = (const float*)d_in[2 + 6 * l + 3];
        const float* mean  = (const float*)d_in[2 + 6 * l + 4];
        const float* var   = (const float*)d_in[2 + 6 * l + 5];
        int C = CO[l];
        sbprep<<<(C + 255) / 256, 256, 0, stream>>>(b, gamma, beta, mean, var,
                                                    scl[l], bia[l], C);
        if (l >= 1) {
            int total = CO[l] * CI[l] * 9;
            wprep<<<(total + 255) / 256, 256, 0, stream>>>(w, wt[l], CI[l],
                                                           CO[l], total);
        }
    }

    // ---- backbone ----
    conv0_direct<<<(256 * 256 * 64) / 256, 256, 0, stream>>>(
        img, actA, (const float*)d_in[2], scl[0], bia[0]);

    ushort_t* cur = actA;
    ushort_t* nxt = actB;
    int Hc = 256;
    for (int l = 1; l < 13; ++l) {
        // zero the halo of the consumer's input buffer (geometry Hc, CI[l])
        {
            int cg = CI[l] >> 3;
            int total = (2 * (Hc + 2) + 2 * Hc) * cg;
            zero_halo<<<(total + 255) / 256, 256, 0, stream>>>(cur, Hc, Hc,
                                                               CI[l]);
        }
        dim3 grid((Hc >> 2) * (Hc >> 4), CO[l] >> 6);
        conv3x3_wmma_bf16<<<grid, 256, 0, stream>>>(cur, nxt, wt[l], scl[l],
                                                    bia[l], Hc, Hc, CI[l],
                                                    CO[l]);
        ushort_t* t = cur; cur = nxt; nxt = t;
        if (l == 1 || l == 3 || l == 6 || l == 9) {
            int H2 = Hc / 2;
            int total = H2 * H2 * CO[l] / 4;
            maxpool2x2<<<(total + 255) / 256, 256, 0, stream>>>(cur, nxt, H2,
                                                                CO[l]);
            t = cur; cur = nxt; nxt = t;
            Hc = H2;
        }
    }

    // ---- ROI pool ----
    roi_pool_k<<<128, 256, 0, stream>>>(cur, rois, (float*)d_out);
}
